// PointerGenerator_979252543895
// MI455X (gfx1250) — matrix operand order repaired and checked by reading, not compiled, not run on hardware
//
#include <hip/hip_runtime.h>
#include <hip/hip_bf16.h>
#include <math.h>

#define B_    4
#define T_    512
#define H_    512
#define V_    32000
#define S_    512
#define SDV_  300
#define TDV_  512
#define ROWS  (B_ * T_)                 // 2048
#define OUT_W (V_ + SDV_ + TDV_)        // 32812
#define PROBS_TOTAL ((size_t)ROWS * OUT_W)

typedef __attribute__((ext_vector_type(16))) __bf16      v16bf;
typedef __attribute__((ext_vector_type(8)))  float       v8f;
typedef __attribute__((ext_vector_type(4)))  unsigned int u32x4;
typedef __attribute__((ext_vector_type(8)))  unsigned int u32x8;

// hardware RNE convert (v_cvt_*bf16_f32) instead of manual bit twiddling
__device__ __forceinline__ unsigned short f2bf(float f) {
    __bf16 h = (__bf16)f;
    return __builtin_bit_cast(unsigned short, h);
}
__device__ __forceinline__ unsigned pk2bf(float lo, float hi) {
    return (unsigned)f2bf(lo) | ((unsigned)f2bf(hi) << 16);
}

// ---------------------------------------------------------------- gate head
__global__ void gates_kernel(const float* __restrict__ hiddens,
                             const float* __restrict__ Wp,
                             const float* __restrict__ bp,
                             float* __restrict__ p_out) {
    int row = blockIdx.x * blockDim.x + threadIdx.x;
    if (row >= ROWS) return;
    const float* h = hiddens + (size_t)row * H_;
    float a0 = bp[0], a1 = bp[1], a2 = bp[2];
    for (int k = 0; k < H_; ++k) {
        float hv = h[k];
        a0 += hv * Wp[k * 3 + 0];
        a1 += hv * Wp[k * 3 + 1];
        a2 += hv * Wp[k * 3 + 2];
    }
    float m  = fmaxf(a0, fmaxf(a1, a2));
    float e0 = __expf(a0 - m), e1 = __expf(a1 - m), e2 = __expf(a2 - m);
    float inv = 1.0f / (e0 + e1 + e2);
    p_out[row * 3 + 0] = e0 * inv;   // p_copy_source
    p_out[row * 3 + 1] = e1 * inv;   // p_copy_target
    p_out[row * 3 + 2] = e2 * inv;   // p_generate
}

// -------------------------------------------- recover indices from one-hots
__global__ void recover_idx_kernel(const float* __restrict__ src_map,  // (B,S,SDV)
                                   const float* __restrict__ tgt_map,  // (B,T,TDV)
                                   int* __restrict__ src_idx,
                                   int* __restrict__ tgt_idx) {
    int r = blockIdx.x * blockDim.x + threadIdx.x;
    if (r >= ROWS) return;
    {
        const float* m = src_map + (size_t)r * SDV_;
        int idx = 0;
        for (int v = 0; v < SDV_; ++v) if (m[v] > 0.5f) idx = v;
        src_idx[r] = idx;
    }
    {
        const float* m = tgt_map + (size_t)r * TDV_;
        int idx = 0;
        for (int v = 0; v < TDV_; ++v) if (m[v] > 0.5f) idx = v;
        tgt_idx[r] = idx;
    }
}

// ------------------------------------------------- WMMA bf16 GEMM: h @ Wv + bv
#define TM 64
#define TN 128
#define TK 32
#define KSTEPS (H_ / TK)   // 16

__global__ __launch_bounds__(256) void score_gemm_kernel(
        const float* __restrict__ A,    // (ROWS, H) hiddens
        const float* __restrict__ Bw,   // (H, V)    Wv
        const float* __restrict__ bv,   // (V)
        float* __restrict__ out)        // (ROWS, OUT_W), vocab slice
{
    __shared__ __align__(16) unsigned short As[TM * TK];  // [m][k] bf16 bits
    __shared__ __align__(16) unsigned short Bs[TN * TK];  // [n][k] bf16 bits (transposed)

    const int tid  = threadIdx.x;
    const int lane = tid & 31;
    const int wid  = tid >> 5;        // 0..7
    const int wm   = wid & 3;         // M subtile (4 x 16)
    const int wn   = wid >> 2;        // N subtile (2 x 64)
    const int n0   = blockIdx.x * TN;
    const int m0   = blockIdx.y * TM;

    v8f acc[4];
#pragma unroll
    for (int j = 0; j < 4; ++j)
#pragma unroll
        for (int r = 0; r < 8; ++r) acc[j][r] = 0.0f;

    // cooperative tile-load coordinates
    const int am  = tid >> 2;          // 0..63   (A row)
    const int ak  = (tid & 3) * 8;     // 0,8,16,24
    const int bk2 = (tid >> 4) * 2;    // 0,2,..,30 (B rows k, k+1)
    const int bn8 = (tid & 15) * 8;    // 0..120    (8 consecutive n)

    const float* gA = A  + (size_t)(m0 + am) * H_ + ak;    // += TK per step
    const float* gB = Bw + (size_t)bk2 * V_ + n0 + bn8;    // += TK*V_ per step

    // fragment LDS addresses (ISA 16-bit layouts)
    const int afm = wm * 16 + (lane & 15);
    const int akb = (lane >> 4) * 8;            // A: lanes 0-15 K 0..7/16..23, 16-31 +8
    const unsigned short* ap0 = &As[afm * TK + akb];
    const unsigned short* ap1 = ap0 + 16;
    const int bkb = (lane >> 4) * 8;            // B: lanes 0-15 K 0..7/16..23, 16-31 +8
    const int nf0 = wn * 64 + (lane & 15);

    // ---- prologue: fetch K-tile 0 into registers
    float4 ra0 = *(const float4*)(gA);
    float4 ra1 = *(const float4*)(gA + 4);
    float4 rb0 = *(const float4*)(gB);            // row k,   n8..n8+3
    float4 rb1 = *(const float4*)(gB + 4);        // row k,   n8+4..n8+7
    float4 rb2 = *(const float4*)(gB + V_);       // row k+1, n8..n8+3
    float4 rb3 = *(const float4*)(gB + V_ + 4);   // row k+1, n8+4..n8+7

    for (int kt = 0; kt < KSTEPS; ++kt) {
        // ---- stage current registers into LDS (fp32 -> bf16, B transposed)
        {
            u32x4 pk;
            pk[0] = pk2bf(ra0.x, ra0.y);
            pk[1] = pk2bf(ra0.z, ra0.w);
            pk[2] = pk2bf(ra1.x, ra1.y);
            pk[3] = pk2bf(ra1.z, ra1.w);
            *(u32x4*)&As[am * TK + ak] = pk;

            // Bs[n][k]: pack (k, k+1) bf16 pair into one dword store
            *(unsigned*)&Bs[(bn8 + 0) * TK + bk2] = pk2bf(rb0.x, rb2.x);
            *(unsigned*)&Bs[(bn8 + 1) * TK + bk2] = pk2bf(rb0.y, rb2.y);
            *(unsigned*)&Bs[(bn8 + 2) * TK + bk2] = pk2bf(rb0.z, rb2.z);
            *(unsigned*)&Bs[(bn8 + 3) * TK + bk2] = pk2bf(rb0.w, rb2.w);
            *(unsigned*)&Bs[(bn8 + 4) * TK + bk2] = pk2bf(rb1.x, rb3.x);
            *(unsigned*)&Bs[(bn8 + 5) * TK + bk2] = pk2bf(rb1.y, rb3.y);
            *(unsigned*)&Bs[(bn8 + 6) * TK + bk2] = pk2bf(rb1.z, rb3.z);
            *(unsigned*)&Bs[(bn8 + 7) * TK + bk2] = pk2bf(rb1.w, rb3.w);
        }
        __syncthreads();

        // ---- software pipeline: launch next K-tile global loads now
        if (kt + 1 < KSTEPS) {
            gA += TK;
            gB += (size_t)TK * V_;
            __builtin_prefetch(gB + (size_t)TK * V_, 0, 1);  // 2 steps ahead -> L2 warm
            ra0 = *(const float4*)(gA);
            ra1 = *(const float4*)(gA + 4);
            rb0 = *(const float4*)(gB);
            rb1 = *(const float4*)(gB + 4);
            rb2 = *(const float4*)(gB + V_);
            rb3 = *(const float4*)(gB + V_ + 4);
        }

        // ---- load ALL fragments first (ds loads overlap), then WMMAs back-to-back
        u32x8 aq;
        {
            u32x4 lo = *(const u32x4*)ap0;
            u32x4 hi = *(const u32x4*)ap1;
            aq[0] = lo[0]; aq[1] = lo[1]; aq[2] = lo[2]; aq[3] = lo[3];
            aq[4] = hi[0]; aq[5] = hi[1]; aq[6] = hi[2]; aq[7] = hi[3];
        }
        v16bf afrag = __builtin_bit_cast(v16bf, aq);

        v16bf bfrag[4];
#pragma unroll
        for (int j = 0; j < 4; ++j) {
            const unsigned short* bp = &Bs[(nf0 + j * 16) * TK + bkb];
            u32x4 lo = *(const u32x4*)bp;
            u32x4 hi = *(const u32x4*)(bp + 16);
            u32x8 bq;
            bq[0] = lo[0]; bq[1] = lo[1]; bq[2] = lo[2]; bq[3] = lo[3];
            bq[4] = hi[0]; bq[5] = hi[1]; bq[6] = hi[2]; bq[7] = hi[3];
            bfrag[j] = __builtin_bit_cast(v16bf, bq);
        }

#pragma unroll
        for (int j = 0; j < 4; ++j)
            acc[j] = __builtin_amdgcn_wmma_f32_16x16x32_bf16(
                         false, afrag, false, bfrag[j], (short)0, acc[j], false, false);

        __syncthreads();
    }

    // epilogue: C layout — vgpr r: lanes 0-15 M=r, lanes 16-31 M=8+r; N=lane%16
#pragma unroll
    for (int j = 0; j < 4; ++j) {
        const int n     = n0 + wn * 64 + j * 16 + (lane & 15);
        const float bvn = bv[n];
        const int mbase = m0 + wm * 16 + ((lane >> 4) << 3);
#pragma unroll
        for (int r = 0; r < 8; ++r)
            out[(size_t)(mbase + r) * OUT_W + n] = acc[j][r] + bvn;
    }
}

// ------------------------------------------ in-place vocab softmax * p_gen
__global__ __launch_bounds__(256) void vocab_softmax_kernel(float* __restrict__ out,
                                                            const float* __restrict__ p_gates) {
    __shared__ float red[256];
    const int row = blockIdx.x;
    float* ptr = out + (size_t)row * OUT_W;
    const int tid = threadIdx.x;

    float lmax = -3.402823466e38f;
    for (int i = tid; i < V_; i += 256) if (i != 0) lmax = fmaxf(lmax, ptr[i]);
    red[tid] = lmax; __syncthreads();
    for (int s = 128; s > 0; s >>= 1) {
        if (tid < s) red[tid] = fmaxf(red[tid], red[tid + s]);
        __syncthreads();
    }
    const float rmax = red[0]; __syncthreads();

    float lsum = 0.f;
    for (int i = tid; i < V_; i += 256) if (i != 0) lsum += __expf(ptr[i] - rmax);
    red[tid] = lsum; __syncthreads();
    for (int s = 128; s > 0; s >>= 1) {
        if (tid < s) red[tid] += red[tid + s];
        __syncthreads();
    }
    const float scale = p_gates[row * 3 + 2] / red[0];
    __syncthreads();

    for (int i = tid; i < V_; i += 256)
        ptr[i] = (i == 0) ? 0.0f : __expf(ptr[i] - rmax) * scale;
}

// -------------------------------------------------- zero copy-prob region
__global__ void zero_copy_kernel(float* __restrict__ out) {
    int id = blockIdx.x * blockDim.x + threadIdx.x;
    const int W = SDV_ + TDV_;
    if (id < ROWS * W) {
        int row = id / W, j = id % W;
        out[(size_t)row * OUT_W + V_ + j] = 0.0f;
    }
}

// ------------------------------------- scatter attention mass (one-hot einsums)
__global__ __launch_bounds__(256) void scatter_kernel(const float* __restrict__ src_att,
                                                      const float* __restrict__ tgt_att,
                                                      const int*   __restrict__ src_idx,
                                                      const int*   __restrict__ tgt_idx,
                                                      const float* __restrict__ p_gates,
                                                      float* __restrict__ out) {
    const int row = blockIdx.x;            // b*T + t
    const int b   = row / T_;
    const float ps = p_gates[row * 3 + 0];
    const float pt = p_gates[row * 3 + 1];
    float* rowp = out + (size_t)row * OUT_W;

    for (int s = threadIdx.x; s < S_; s += 256) {
        float a = src_att[(size_t)row * S_ + s] * ps;
        atomicAdd(&rowp[V_ + src_idx[b * S_ + s]], a);
    }
    for (int n = threadIdx.x; n < TDV_; n += 256) {
        float a = tgt_att[(size_t)row * TDV_ + n] * pt;
        atomicAdd(&rowp[V_ + SDV_ + tgt_idx[b * T_ + n]], a);
    }
}

// --------------------------------------------------------------- argmax row
__global__ __launch_bounds__(256) void argmax_kernel(const float* __restrict__ probs,
                                                     float* __restrict__ pred) {
    __shared__ float rv[256];
    __shared__ int   ri[256];
    const int row = blockIdx.x;
    const float* ptr = probs + (size_t)row * OUT_W;
    float bval = -3.402823466e38f; int bidx = 0;
    for (int i = threadIdx.x; i < OUT_W; i += 256) {
        float v = ptr[i];
        if (v > bval || (v == bval && i < bidx)) { bval = v; bidx = i; }
    }
    rv[threadIdx.x] = bval; ri[threadIdx.x] = bidx; __syncthreads();
    for (int s = 128; s > 0; s >>= 1) {
        if (threadIdx.x < s) {
            float ov = rv[threadIdx.x + s]; int oi = ri[threadIdx.x + s];
            if (ov > rv[threadIdx.x] || (ov == rv[threadIdx.x] && oi < ri[threadIdx.x])) {
                rv[threadIdx.x] = ov; ri[threadIdx.x] = oi;
            }
        }
        __syncthreads();
    }
    if (threadIdx.x == 0) pred[row] = (float)ri[0];
}

// --------------------------------------------------------------------- host
extern "C" void kernel_launch(void* const* d_in, const int* in_sizes, int n_in,
                              void* d_out, int out_size, void* d_ws, size_t ws_size,
                              hipStream_t stream) {
    (void)in_sizes; (void)n_in; (void)out_size; (void)ws_size;
    const float* hiddens = (const float*)d_in[0];
    const float* Wp      = (const float*)d_in[1];
    const float* bp      = (const float*)d_in[2];
    const float* Wv      = (const float*)d_in[3];
    const float* bv      = (const float*)d_in[4];
    const float* src_att = (const float*)d_in[5];
    const float* src_map = (const float*)d_in[6];
    const float* tgt_att = (const float*)d_in[7];
    const float* tgt_map = (const float*)d_in[8];
    float* out = (float*)d_out;

    float* p_gates = (float*)d_ws;                                   // 2048*3 f32
    int*   src_idx = (int*)((char*)d_ws + ROWS * 3 * sizeof(float)); // 2048 i32
    int*   tgt_idx = src_idx + ROWS;                                 // 2048 i32

    gates_kernel<<<ROWS / 256, 256, 0, stream>>>(hiddens, Wp, bp, p_gates);
    recover_idx_kernel<<<ROWS / 256, 256, 0, stream>>>(src_map, tgt_map, src_idx, tgt_idx);

    dim3 ggrid(V_ / TN, ROWS / TM);   // (250, 32)
    score_gemm_kernel<<<ggrid, 256, 0, stream>>>(hiddens, Wv, bv, out);

    vocab_softmax_kernel<<<ROWS, 256, 0, stream>>>(out, p_gates);
    zero_copy_kernel<<<(ROWS * (SDV_ + TDV_) + 255) / 256, 256, 0, stream>>>(out);
    scatter_kernel<<<ROWS, 256, 0, stream>>>(src_att, tgt_att, src_idx, tgt_idx, p_gates, out);
    argmax_kernel<<<ROWS, 256, 0, stream>>>(out, out + PROBS_TOTAL);
}
